// GNN_68719476736453
// MI455X (gfx1250) — matrix operand order repaired and checked by reading, not compiled, not run on hardware
//
#include <hip/hip_runtime.h>
#include <hip/hip_bf16.h>

// ---------------------------------------------------------------------------
// MI455X (gfx1250) GraphNet. wave32, WMMA bf16 (f32 accum).
// meta_node kept bf16 (33.5MB -> L2 resident) so edge gathers hit L2.
// meta_edge bf16 streamed. Weights staged per-block in LDS, pre-packed in
// B-fragment order (one 32B ds load per 32x16 fragment).
// ---------------------------------------------------------------------------

#define NN      262144   // nodes
#define NE      2097152  // edges
#define BATCH   1024
#define HID     64
#define FFD     32
#define EDIM    20

typedef __attribute__((ext_vector_type(16))) __bf16 v16bf;
typedef __attribute__((ext_vector_type(8)))  __bf16 v8bf;
typedef __attribute__((ext_vector_type(8)))  float  v8f;

__device__ __forceinline__ v8f wmma_bf16(v16bf a, v16bf b, v8f c) {
    // 8 args: (neg_a, A, neg_b, B, c_mod, C, reuse_a, reuse_b)
    return __builtin_amdgcn_wmma_f32_16x16x32_bf16(false, a, false, b,
                                                   (short)0, c, false, false);
}

// ---- A fragment (16x32 bf16, K-chunk kc) from a bf16 row-major row --------
// ISA layout: lane<16 -> M=lane, K in {kb..kb+7} and {16+kb..}; kb = 8*(lane>=16)
__device__ __forceinline__ v16bf a_frag_bf16row(const __bf16* __restrict__ rowp, int kc) {
    const int lane = threadIdx.x & 31;
    const int kb = kc * 32 + ((lane >> 4) << 3);
    v8bf lo = *(const v8bf*)(rowp + kb);
    v8bf hi = *(const v8bf*)(rowp + kb + 16);
    v16bf a;
#pragma unroll
    for (int i = 0; i < 8; ++i) { a[i] = lo[i]; a[i + 8] = hi[i]; }
    return a;
}

__device__ __forceinline__ v16bf a_frag_f32row(const float* __restrict__ rowp, int kc) {
    const int lane = threadIdx.x & 31;
    const int kb = kc * 32 + ((lane >> 4) << 3);
    v16bf a;
#pragma unroll
    for (int i = 0; i < 8; ++i) {
        a[i]     = (__bf16)rowp[kb + i];
        a[i + 8] = (__bf16)rowp[kb + 16 + i];
    }
    return a;
}

// padded f32 row (edge features: 20 valid K, rest 0)
__device__ __forceinline__ v16bf a_frag_f32row_pad(const float* __restrict__ rowp, int kvalid) {
    const int lane = threadIdx.x & 31;
    const int kb = (lane >> 4) << 3;
    v16bf a;
#pragma unroll
    for (int i = 0; i < 8; ++i) {
        int k0 = kb + i, k1 = kb + 16 + i;
        a[i]     = (__bf16)(k0 < kvalid ? rowp[k0] : 0.f);
        a[i + 8] = (__bf16)(k1 < kvalid ? rowp[k1] : 0.f);
    }
    return a;
}

// ---- B fragments: stage weight [K x N] f32 -> LDS pre-packed per-fragment --
// frag f = kc*NT + nt; element index = (f*32 + lane)*16 + j
// lane: n = nt*16 + (lane&15), ko = kc*32 + 8*(lane>>4); b[j] = W[(ko+(j&7)+16*(j>>3))*N + n]
__device__ __forceinline__ void stage_w_frag(__bf16* dst, const float* __restrict__ W,
                                             int K, int N, int kvalid, int tid, int nthr) {
    const int NT = N >> 4;
    const int total = K * N;
    for (int i = tid; i < total; i += nthr) {
        int j = i & 15;
        int lane = (i >> 4) & 31;
        int f = i >> 9;
        int nt = f % NT, kc = f / NT;
        int n = nt * 16 + (lane & 15);
        int k = kc * 32 + ((lane >> 4) << 3) + (j & 7) + ((j >> 3) << 4);
        dst[i] = (__bf16)(k < kvalid ? W[k * N + n] : 0.f);
    }
}

__device__ __forceinline__ v16bf b_frag(const __bf16* __restrict__ Wp, int NT, int kc, int nt) {
    const int lane = threadIdx.x & 31;
    const int f = kc * NT + nt;
    return *(const v16bf*)(Wp + (((f << 5) + lane) << 4));
}

__device__ __forceinline__ void stage_f(float* dst, const float* __restrict__ src,
                                        int n, int tid, int nthr) {
    for (int i = tid; i < n; i += nthr) dst[i] = src[i];
}

// C layout: VGPR r holds (m = r + 8*(lane>>4), n = nt*16 + (lane&15))
__device__ __forceinline__ void c_to_lds_bf16(__bf16* dst, int ld, int nt, v8f acc) {
    const int lane = threadIdx.x & 31;
    const int n = nt * 16 + (lane & 15);
    const int mb = (lane >> 4) << 3;
#pragma unroll
    for (int r = 0; r < 8; ++r) dst[(mb + r) * ld + n] = (__bf16)acc[r];
}

// ===========================================================================
// Global-state MLP (tiny, VALU): tot = relu(gs@W1+b1)@W2+b2 + gs@linW + linb
// ===========================================================================
__global__ void __launch_bounds__(256) k_state(
    const float* __restrict__ gs, const float* __restrict__ W1, const float* __restrict__ b1,
    const float* __restrict__ W2, const float* __restrict__ b2,
    const float* __restrict__ linW, const float* __restrict__ linb,
    float* __restrict__ tot) {
    __shared__ float h[FFD];
    const int tid = threadIdx.x;
    const int j = blockIdx.x * 256 + tid;
    if (tid < FFD) {
        float s = 0.f;
        for (int k = 0; k < BATCH; ++k) s += gs[k] * W1[k * FFD + tid];
        h[tid] = fmaxf(s + b1[tid], 0.f);
    }
    __syncthreads();
    float s = 0.f;
    for (int i = 0; i < FFD; ++i) s += h[i] * W2[i * BATCH + j];
    float t = 0.f;
    for (int k = 0; k < BATCH; ++k) t += gs[k] * linW[k * BATCH + j];
    tot[j] = s + b2[j] + t + linb[j];
}

// ===========================================================================
// Node prep: node_res = relu(x@W1+b1)@W2+b2 ; meta = node_res + x@linW + linb
// grid*4 waves must divide NN/16 (uniform __syncthreads trips)
// ===========================================================================
__global__ void __launch_bounds__(128) k_prep_nodes(
    const float* __restrict__ xf,
    const float* __restrict__ W1, const float* __restrict__ b1,
    const float* __restrict__ W2, const float* __restrict__ b2,
    const float* __restrict__ Wl, const float* __restrict__ bl,
    float* __restrict__ node_res, __bf16* __restrict__ meta_node) {
    __shared__ __bf16 sW1[HID * FFD];   // 64x32
    __shared__ __bf16 sW2[FFD * HID];   // 32x64
    __shared__ __bf16 sWl[HID * HID];   // 64x64
    __shared__ float sb1[FFD], sb2[HID], sbl[HID];
    __shared__ __bf16 sh[4][16 * FFD];
    const int tid = threadIdx.x;
    stage_w_frag(sW1, W1, HID, FFD, HID, tid, 128);
    stage_w_frag(sW2, W2, FFD, HID, FFD, tid, 128);
    stage_w_frag(sWl, Wl, HID, HID, HID, tid, 128);
    stage_f(sb1, b1, FFD, tid, 128);
    stage_f(sb2, b2, HID, tid, 128);
    stage_f(sbl, bl, HID, tid, 128);
    __syncthreads();

    const int wave = tid >> 5, lane = tid & 31;
    __bf16* h = sh[wave];
    const int gw = blockIdx.x * 4 + wave, nw = gridDim.x * 4;
    const int ntiles = NN / 16;
    for (int t = gw; t < ntiles; t += nw) {
        const int m = lane & 15;
        const float* xrow = xf + (size_t)(t * 16 + m) * HID;
        v16bf xa0 = a_frag_f32row(xrow, 0);
        v16bf xa1 = a_frag_f32row(xrow, 1);
#pragma unroll
        for (int nt = 0; nt < 2; ++nt) {                       // hidden [16,32]
            v8f acc = {};
            acc = wmma_bf16(xa0, b_frag(sW1, 2, 0, nt), acc);
            acc = wmma_bf16(xa1, b_frag(sW1, 2, 1, nt), acc);
            const float bb = sb1[nt * 16 + (lane & 15)];
#pragma unroll
            for (int r = 0; r < 8; ++r) acc[r] = fmaxf(acc[r] + bb, 0.f);
            c_to_lds_bf16(h, FFD, nt, acc);
        }
        __syncthreads();
        v16bf ha = a_frag_bf16row(h + m * FFD, 0);
#pragma unroll
        for (int nt = 0; nt < 4; ++nt) {                       // out [16,64]
            v8f rv = {};
            rv = wmma_bf16(ha, b_frag(sW2, 4, 0, nt), rv);
            v8f lv = {};
            lv = wmma_bf16(xa0, b_frag(sWl, 4, 0, nt), lv);
            lv = wmma_bf16(xa1, b_frag(sWl, 4, 1, nt), lv);
            const int n = nt * 16 + (lane & 15);
            const int mb = (lane >> 4) << 3;
            const float b2v = sb2[n], blv = sbl[n];
#pragma unroll
            for (int r = 0; r < 8; ++r) {
                const size_t idx = (size_t)(t * 16 + mb + r) * HID + n;
                const float res = rv[r] + b2v;
                node_res[idx] = res;
                meta_node[idx] = (__bf16)(res + lv[r] + blv);
            }
        }
        __syncthreads();
    }
}

// ===========================================================================
// Edge prep: meta_edge = relu(e@W1+b1)@W2+b2 + e@linW + linb   (K padded 20->32)
// ===========================================================================
__global__ void __launch_bounds__(128) k_prep_edges(
    const float* __restrict__ ef,
    const float* __restrict__ W1, const float* __restrict__ b1,
    const float* __restrict__ W2, const float* __restrict__ b2,
    const float* __restrict__ Wl, const float* __restrict__ bl,
    __bf16* __restrict__ meta_edge) {
    __shared__ __bf16 sW1[32 * FFD];    // padded 32x32
    __shared__ __bf16 sW2[FFD * HID];
    __shared__ __bf16 sWl[32 * HID];    // padded 32x64
    __shared__ float sb1[FFD], sb2[HID], sbl[HID];
    __shared__ __bf16 sh[4][16 * FFD];
    const int tid = threadIdx.x;
    stage_w_frag(sW1, W1, 32, FFD, EDIM, tid, 128);
    stage_w_frag(sW2, W2, FFD, HID, FFD, tid, 128);
    stage_w_frag(sWl, Wl, 32, HID, EDIM, tid, 128);
    stage_f(sb1, b1, FFD, tid, 128);
    stage_f(sb2, b2, HID, tid, 128);
    stage_f(sbl, bl, HID, tid, 128);
    __syncthreads();

    const int wave = tid >> 5, lane = tid & 31;
    __bf16* h = sh[wave];
    const int gw = blockIdx.x * 4 + wave, nw = gridDim.x * 4;
    const int ntiles = NE / 16;
    for (int t = gw; t < ntiles; t += nw) {
        const int m = lane & 15;
        const float* erow = ef + (size_t)(t * 16 + m) * EDIM;
        v16bf ea = a_frag_f32row_pad(erow, EDIM);
#pragma unroll
        for (int nt = 0; nt < 2; ++nt) {
            v8f acc = {};
            acc = wmma_bf16(ea, b_frag(sW1, 2, 0, nt), acc);
            const float bb = sb1[nt * 16 + (lane & 15)];
#pragma unroll
            for (int r = 0; r < 8; ++r) acc[r] = fmaxf(acc[r] + bb, 0.f);
            c_to_lds_bf16(h, FFD, nt, acc);
        }
        __syncthreads();
        v16bf ha = a_frag_bf16row(h + m * FFD, 0);
#pragma unroll
        for (int nt = 0; nt < 4; ++nt) {
            v8f mv = {};
            mv = wmma_bf16(ha, b_frag(sW2, 4, 0, nt), mv);
            v8f lv = {};
            lv = wmma_bf16(ea, b_frag(sWl, 4, 0, nt), lv);
            const int n = nt * 16 + (lane & 15);
            const int mb = (lane >> 4) << 3;
            const float b2v = sb2[n], blv = sbl[n];
#pragma unroll
            for (int r = 0; r < 8; ++r)
                meta_edge[(size_t)(t * 16 + mb + r) * HID + n] =
                    (__bf16)(mv[r] + b2v + lv[r] + blv);
        }
        __syncthreads();
    }
}

// ===========================================================================
// Edge message + scatter-add:
// h=relu(mn[src]@Ws + mn[dst]@Wd + me@We + b1); msg=h@W2+b2; agg[dst]+=msg
// ===========================================================================
__global__ void __launch_bounds__(128) k_edge(
    const int* __restrict__ src, const int* __restrict__ dst,
    const __bf16* __restrict__ mn, const __bf16* __restrict__ me,
    const float* __restrict__ Ws, const float* __restrict__ Wd,
    const float* __restrict__ We, const float* __restrict__ b1,
    const float* __restrict__ W2, const float* __restrict__ b2,
    float* __restrict__ agg) {
    __shared__ __bf16 sWs[HID * FFD], sWd[HID * FFD], sWe[HID * FFD];
    __shared__ __bf16 sW2[FFD * HID];
    __shared__ float sb1[FFD], sb2[HID];
    __shared__ __bf16 sh[4][16 * FFD];
    __shared__ int sdst[4][16];
    const int tid = threadIdx.x;
    stage_w_frag(sWs, Ws, HID, FFD, HID, tid, 128);
    stage_w_frag(sWd, Wd, HID, FFD, HID, tid, 128);
    stage_w_frag(sWe, We, HID, FFD, HID, tid, 128);
    stage_w_frag(sW2, W2, FFD, HID, FFD, tid, 128);
    stage_f(sb1, b1, FFD, tid, 128);
    stage_f(sb2, b2, HID, tid, 128);
    __syncthreads();

    const int wave = tid >> 5, lane = tid & 31;
    __bf16* h = sh[wave];
    const int gw = blockIdx.x * 4 + wave, nw = gridDim.x * 4;
    const int ntiles = NE / 16;
    for (int t = gw; t < ntiles; t += nw) {
        const int m = lane & 15;
        const size_t e = (size_t)t * 16 + m;
        const int si = src[e];
        const int di = dst[e];
        if (lane < 16) sdst[wave][lane] = di;
        const __bf16* srow = mn + (size_t)si * HID;
        const __bf16* drow = mn + (size_t)di * HID;
        const __bf16* erow = me + e * HID;
        v16bf as0 = a_frag_bf16row(srow, 0), as1 = a_frag_bf16row(srow, 1);
        v16bf ad0 = a_frag_bf16row(drow, 0), ad1 = a_frag_bf16row(drow, 1);
        v16bf ae0 = a_frag_bf16row(erow, 0), ae1 = a_frag_bf16row(erow, 1);
#pragma unroll
        for (int nt = 0; nt < 2; ++nt) {                       // h [16,32]
            v8f acc = {};
            acc = wmma_bf16(as0, b_frag(sWs, 2, 0, nt), acc);
            acc = wmma_bf16(as1, b_frag(sWs, 2, 1, nt), acc);
            acc = wmma_bf16(ad0, b_frag(sWd, 2, 0, nt), acc);
            acc = wmma_bf16(ad1, b_frag(sWd, 2, 1, nt), acc);
            acc = wmma_bf16(ae0, b_frag(sWe, 2, 0, nt), acc);
            acc = wmma_bf16(ae1, b_frag(sWe, 2, 1, nt), acc);
            const float bb = sb1[nt * 16 + (lane & 15)];
#pragma unroll
            for (int r = 0; r < 8; ++r) acc[r] = fmaxf(acc[r] + bb, 0.f);
            c_to_lds_bf16(h, FFD, nt, acc);
        }
        __syncthreads();
        v16bf ha = a_frag_bf16row(h + m * FFD, 0);
#pragma unroll
        for (int nt = 0; nt < 4; ++nt) {                       // msg [16,64]
            v8f mv = {};
            mv = wmma_bf16(ha, b_frag(sW2, 4, 0, nt), mv);
            const int n = nt * 16 + (lane & 15);
            const int mb = (lane >> 4) << 3;
            const float bb = sb2[n];
#pragma unroll
            for (int r = 0; r < 8; ++r) {
                const int row = sdst[wave][mb + r];
                atomicAdd(&agg[(size_t)row * HID + n], mv[r] + bb);
            }
        }
        __syncthreads();
    }
}

// ===========================================================================
// Node update:
// hu=relu(mn@Wx+agg@Wagg+b1); upd=hu@W2+b2+state; out=upd+relu(upd@Wr+br);
// meta_node = node_res + out   (also f32 -> d_out on final layer)
// ===========================================================================
__global__ void __launch_bounds__(128) k_update(
    const __bf16* __restrict__ mn_in, const float* __restrict__ agg,
    const float* __restrict__ node_res, const float* __restrict__ tot,
    const int* __restrict__ gszp,
    const float* __restrict__ Wx, const float* __restrict__ Wa,
    const float* __restrict__ b1, const float* __restrict__ W2,
    const float* __restrict__ b2, const float* __restrict__ Wr,
    const float* __restrict__ br,
    __bf16* __restrict__ mn_out, float* __restrict__ outf, int write_out) {
    __shared__ __bf16 sWx[HID * FFD], sWa[HID * FFD];
    __shared__ __bf16 sW2[FFD * HID];
    __shared__ __bf16 sWr[HID * HID];
    __shared__ float sb1[FFD], sb2[HID], sbr[HID];
    __shared__ __bf16 sh[4][16 * FFD];
    __shared__ __bf16 su[4][16 * HID];
    const int tid = threadIdx.x;
    stage_w_frag(sWx, Wx, HID, FFD, HID, tid, 128);
    stage_w_frag(sWa, Wa, HID, FFD, HID, tid, 128);
    stage_w_frag(sW2, W2, FFD, HID, FFD, tid, 128);
    stage_w_frag(sWr, Wr, HID, HID, HID, tid, 128);
    stage_f(sb1, b1, FFD, tid, 128);
    stage_f(sb2, b2, HID, tid, 128);
    stage_f(sbr, br, HID, tid, 128);
    __syncthreads();

    const int wave = tid >> 5, lane = tid & 31;
    __bf16* h = sh[wave];
    __bf16* u = su[wave];
    const int gsz = gszp[0];
    const int gw = blockIdx.x * 4 + wave, nw = gridDim.x * 4;
    const int ntiles = NN / 16;
    for (int t = gw; t < ntiles; t += nw) {
        const int m = lane & 15;
        const int rowbase = t * 16;
        const __bf16* xrow = mn_in + (size_t)(rowbase + m) * HID;
        const float* grow = agg + (size_t)(rowbase + m) * HID;
        v16bf x0 = a_frag_bf16row(xrow, 0), x1 = a_frag_bf16row(xrow, 1);
        v16bf g0 = a_frag_f32row(grow, 0), g1 = a_frag_f32row(grow, 1);
        const float sstate = tot[rowbase / gsz];   // GROUP=256 > tile span
#pragma unroll
        for (int nt = 0; nt < 2; ++nt) {                       // hu [16,32]
            v8f acc = {};
            acc = wmma_bf16(x0, b_frag(sWx, 2, 0, nt), acc);
            acc = wmma_bf16(x1, b_frag(sWx, 2, 1, nt), acc);
            acc = wmma_bf16(g0, b_frag(sWa, 2, 0, nt), acc);
            acc = wmma_bf16(g1, b_frag(sWa, 2, 1, nt), acc);
            const float bb = sb1[nt * 16 + (lane & 15)];
#pragma unroll
            for (int r = 0; r < 8; ++r) acc[r] = fmaxf(acc[r] + bb, 0.f);
            c_to_lds_bf16(h, FFD, nt, acc);
        }
        __syncthreads();
        v16bf ha = a_frag_bf16row(h + m * FFD, 0);
        v8f uacc[4];
#pragma unroll
        for (int nt = 0; nt < 4; ++nt) {                       // upd [16,64]
            v8f uv = {};
            uv = wmma_bf16(ha, b_frag(sW2, 4, 0, nt), uv);
            const float bb = sb2[nt * 16 + (lane & 15)];
#pragma unroll
            for (int r = 0; r < 8; ++r) uv[r] += bb + sstate;
            uacc[nt] = uv;
            c_to_lds_bf16(u, HID, nt, uv);
        }
        __syncthreads();
        v16bf u0 = a_frag_bf16row(u + m * HID, 0);
        v16bf u1 = a_frag_bf16row(u + m * HID, 1);
#pragma unroll
        for (int nt = 0; nt < 4; ++nt) {                       // residual block
            v8f rv = {};
            rv = wmma_bf16(u0, b_frag(sWr, 4, 0, nt), rv);
            rv = wmma_bf16(u1, b_frag(sWr, 4, 1, nt), rv);
            const int n = nt * 16 + (lane & 15);
            const int mb = (lane >> 4) << 3;
            const float bb = sbr[n];
#pragma unroll
            for (int r = 0; r < 8; ++r) {
                const size_t idx = (size_t)(rowbase + mb + r) * HID + n;
                const float out = uacc[nt][r] + fmaxf(rv[r] + bb, 0.f);
                const float nm = node_res[idx] + out;
                mn_out[idx] = (__bf16)nm;
                if (write_out) outf[idx] = nm;
            }
        }
        __syncthreads();
    }
}

// ===========================================================================
extern "C" void kernel_launch(void* const* d_in, const int* in_sizes, int n_in,
                              void* d_out, int out_size, void* d_ws, size_t ws_size,
                              hipStream_t stream) {
    (void)in_sizes; (void)n_in; (void)out_size; (void)ws_size;
    const float* node_feature = (const float*)d_in[0];
    const int*   edge_index   = (const int*)d_in[1];
    const float* edge_feature = (const float*)d_in[2];
    const float* global_state = (const float*)d_in[3];
    const int*   group_size   = (const int*)d_in[4];
    const float* node_lin_W = (const float*)d_in[5];
    const float* node_lin_b = (const float*)d_in[6];
    const float* edge_lin_W = (const float*)d_in[7];
    const float* edge_lin_b = (const float*)d_in[8];
    const float* state_lin_W = (const float*)d_in[9];
    const float* state_lin_b = (const float*)d_in[10];
    const float* node_ff_W1 = (const float*)d_in[11];
    const float* node_ff_b1 = (const float*)d_in[12];
    const float* node_ff_W2 = (const float*)d_in[13];
    const float* node_ff_b2 = (const float*)d_in[14];
    const float* edge_ff_W1 = (const float*)d_in[15];
    const float* edge_ff_b1 = (const float*)d_in[16];
    const float* edge_ff_W2 = (const float*)d_in[17];
    const float* edge_ff_b2 = (const float*)d_in[18];
    const float* state_ff_W1 = (const float*)d_in[19];
    const float* state_ff_b1 = (const float*)d_in[20];
    const float* state_ff_W2 = (const float*)d_in[21];
    const float* state_ff_b2 = (const float*)d_in[22];
    const float* cWsrc = (const float*)d_in[23];
    const float* cWdst = (const float*)d_in[24];
    const float* cWe   = (const float*)d_in[25];
    const float* cb1   = (const float*)d_in[26];
    const float* cW2   = (const float*)d_in[27];
    const float* cb2   = (const float*)d_in[28];
    const float* uWx   = (const float*)d_in[29];
    const float* uWagg = (const float*)d_in[30];
    const float* ub1   = (const float*)d_in[31];
    const float* uW2   = (const float*)d_in[32];
    const float* ub2   = (const float*)d_in[33];
    const float* rW    = (const float*)d_in[34];
    const float* rb    = (const float*)d_in[35];

    float* outp = (float*)d_out;

    // workspace partition (256B aligned slabs)
    char* ws = (char*)d_ws;
    __bf16* meta_node = (__bf16*)ws; ws += (size_t)NN * HID * sizeof(__bf16);
    float*  node_res  = (float*)ws;  ws += (size_t)NN * HID * sizeof(float);
    __bf16* meta_edge = (__bf16*)ws; ws += (size_t)NE * HID * sizeof(__bf16);
    float*  agg       = (float*)ws;  ws += (size_t)NN * HID * sizeof(float);
    float*  tot       = (float*)ws;  ws += (size_t)BATCH * sizeof(float);

    const int* src = edge_index;
    const int* dst = edge_index + NE;

    k_state<<<BATCH / 256, 256, 0, stream>>>(global_state, state_ff_W1, state_ff_b1,
                                             state_ff_W2, state_ff_b2,
                                             state_lin_W, state_lin_b, tot);
    // 1024 blocks * 4 waves = 4096 waves -> 16384 node tiles / 4096 = 4 (uniform)
    k_prep_nodes<<<1024, 128, 0, stream>>>(node_feature, node_ff_W1, node_ff_b1,
                                           node_ff_W2, node_ff_b2,
                                           node_lin_W, node_lin_b,
                                           node_res, meta_node);
    // 4096 blocks * 4 waves = 16384 waves -> 131072 edge tiles / 16384 = 8 (uniform)
    k_prep_edges<<<4096, 128, 0, stream>>>(edge_feature, edge_ff_W1, edge_ff_b1,
                                           edge_ff_W2, edge_ff_b2,
                                           edge_lin_W, edge_lin_b, meta_edge);

    for (int l = 0; l < 3; ++l) {
        hipMemsetAsync(agg, 0, (size_t)NN * HID * sizeof(float), stream);
        k_edge<<<4096, 128, 0, stream>>>(src, dst, meta_node, meta_edge,
                                         cWsrc + (size_t)l * HID * FFD,
                                         cWdst + (size_t)l * HID * FFD,
                                         cWe   + (size_t)l * HID * FFD,
                                         cb1   + (size_t)l * FFD,
                                         cW2   + (size_t)l * FFD * HID,
                                         cb2   + (size_t)l * HID,
                                         agg);
        k_update<<<1024, 128, 0, stream>>>(meta_node, agg, node_res, tot, group_size,
                                           uWx   + (size_t)l * HID * FFD,
                                           uWagg + (size_t)l * HID * FFD,
                                           ub1   + (size_t)l * FFD,
                                           uW2   + (size_t)l * FFD * HID,
                                           ub2   + (size_t)l * HID,
                                           rW    + (size_t)l * HID * HID,
                                           rb    + (size_t)l * HID,
                                           meta_node, outp, (l == 2) ? 1 : 0);
    }
}
